// EfficientMambaAttention_33904471835297
// MI455X (gfx1250) — matrix operand order repaired and checked by reading, not compiled, not run on hardware
//
#include <hip/hip_runtime.h>
#include <hip/hip_bf16.h>
#include <math.h>

// ---------------- problem constants ----------------
#define BB   8
#define CC_  128
#define HH   160
#define WW   160
#define LL   160
#define DIN  256
#define DST  16
#define DR   8

typedef __bf16 bf16_t;
typedef __attribute__((ext_vector_type(16))) __bf16 v16bf_t;
typedef __attribute__((ext_vector_type(8)))  float  v8f_t;

__device__ __forceinline__ float sigmoidf_(float x){ return 1.f/(1.f+__expf(-x)); }
__device__ __forceinline__ float siluf_(float x){ return x*sigmoidf_(x); }
__device__ __forceinline__ float geluf_(float x){ return 0.5f*x*(1.f+erff(x*0.70710678118654752f)); }
__device__ __forceinline__ float softplusf_(float x){ return (x>20.f)?x:log1pf(__expf(x)); }

// ---------------------------------------------------------------------------
// Kernel 1: per-(b,c) plane row-means and col-means in one pass over x.
// block = 160 threads (5 waves); wave w owns rows [32w,32w+32); lane owns 5 cols.
// ---------------------------------------------------------------------------
__global__ void __launch_bounds__(160) k_means(const float* __restrict__ x,
                                               float* __restrict__ rowmean,
                                               float* __restrict__ colmean){
  int bc = blockIdx.x;                       // b*128 + c
  const float* xp = x + (size_t)bc*HH*WW;
  int tid = threadIdx.x, wv = tid>>5, lane = tid&31;
  __shared__ float colpart[5][WW];
  float cacc[5] = {0.f,0.f,0.f,0.f,0.f};
  for (int r=0;r<32;r++){
    int h = wv*32 + r;
    float rs = 0.f;
    #pragma unroll
    for (int k=0;k<5;k++){
      float v = xp[h*WW + k*32 + lane];
      cacc[k]+=v; rs+=v;
    }
    #pragma unroll
    for (int off=16; off; off>>=1) rs += __shfl_down(rs, off, 32);
    if (lane==0) rowmean[(size_t)bc*HH + h] = rs*(1.f/(float)WW);
  }
  #pragma unroll
  for (int k=0;k<5;k++) colpart[wv][k*32+lane] = cacc[k];
  __syncthreads();
  if (tid<WW){
    float s=0.f;
    #pragma unroll
    for (int w2=0;w2<5;w2++) s+=colpart[w2][tid];
    colmean[(size_t)bc*WW+tid] = s*(1.f/(float)HH);
  }
}

// ---------------------------------------------------------------------------
// Kernel 2: y = gelu(bn(conv1(concat(x_h,x_w)))) then a_h / a_w (8->128).
// grid = B*320 positions, block = 128.
// ---------------------------------------------------------------------------
__global__ void __launch_bounds__(128) k_y_ahw(const float* __restrict__ rowmean,
    const float* __restrict__ colmean,
    const float* __restrict__ conv1_w, const float* __restrict__ conv1_b,
    const float* __restrict__ bn_g, const float* __restrict__ bn_b,
    const float* __restrict__ bn_m, const float* __restrict__ bn_v,
    const float* __restrict__ convh_w, const float* __restrict__ convh_b,
    const float* __restrict__ convw_w, const float* __restrict__ convw_b,
    float* __restrict__ a_h, float* __restrict__ a_w){
  int blk = blockIdx.x; int b = blk/(HH+WW); int pos = blk%(HH+WW);
  int t = threadIdx.x;
  __shared__ float mv[CC_];
  __shared__ float ys[8];
  int ish = (pos<HH)?1:0;
  int p = ish ? pos : (pos-HH);
  const float* src = ish ? rowmean : colmean;
  mv[t] = src[((size_t)b*CC_ + t)*HH + p];
  __syncthreads();
  if (t<8){
    float a = conv1_b[t];
    for (int d=0; d<CC_; d++) a += conv1_w[t*CC_+d]*mv[d];
    a = (a - bn_m[t])*rsqrtf(bn_v[t]+1e-5f)*bn_g[t] + bn_b[t];
    ys[t] = geluf_(a);
  }
  __syncthreads();
  const float* wgt = ish ? convh_w : convw_w;
  const float* bia = ish ? convh_b : convw_b;
  float a = bia[t];
  #pragma unroll
  for (int j=0;j<8;j++) a += wgt[t*8+j]*ys[j];
  float* dst = ish ? a_h : a_w;
  dst[((size_t)b*CC_+t)*HH + p] = a;
}

// ---------------------------------------------------------------------------
// Kernel 3: xz[b,l,e] = sum_d u[b,l,d] * in_w[e,d]   (e < 512)
// ---------------------------------------------------------------------------
__global__ void __launch_bounds__(256) k_inproj(const float* __restrict__ mean,
                                                const float* __restrict__ in_w,
                                                float* __restrict__ xz){
  int blk = blockIdx.x; int b = blk/LL, l = blk%LL;
  int t = threadIdx.x;
  __shared__ float uv[CC_];
  if (t<CC_) uv[t] = mean[((size_t)b*CC_+t)*LL + l];
  __syncthreads();
  float* o = xz + (size_t)blk*(2*DIN);
  #pragma unroll
  for (int half=0; half<2; half++){
    int e = t + half*256;
    float a = 0.f;
    for (int d=0;d<CC_;d++) a += in_w[(size_t)e*CC_+d]*uv[d];
    o[e] = a;
  }
}

// ---------------------------------------------------------------------------
// Kernel 4: causal depthwise conv (d_conv=4) + SiLU
// ---------------------------------------------------------------------------
__global__ void __launch_bounds__(256) k_convsilu(const float* __restrict__ xz,
                                                  const float* __restrict__ conv_w,
                                                  const float* __restrict__ conv_b,
                                                  float* __restrict__ xc){
  int blk=blockIdx.x; int b=blk/LL, l=blk%LL; int d=threadIdx.x;
  float a = conv_b[d];
  #pragma unroll
  for (int k=0;k<4;k++){
    int ll = l-3+k;
    if (ll>=0) a += conv_w[d*4+k]*xz[((size_t)b*LL+ll)*(2*DIN)+d];
  }
  xc[(size_t)blk*DIN + d] = siluf_(a);
}

// ---------------------------------------------------------------------------
// Kernel 5: dbc = xc @ xproj^T ; delta = softplus(dt@dt_w^T + dt_b); split B_,C_
// ---------------------------------------------------------------------------
__global__ void __launch_bounds__(256) k_xproj_delta(const float* __restrict__ xc,
    const float* __restrict__ xproj_w, const float* __restrict__ dt_w,
    const float* __restrict__ dt_b,
    float* __restrict__ delta, float* __restrict__ Bc, float* __restrict__ Cc){
  int blk=blockIdx.x; int t=threadIdx.x;
  __shared__ float xv[DIN];
  __shared__ float dbc[DR+2*DST];
  xv[t] = xc[(size_t)blk*DIN+t];
  __syncthreads();
  if (t<(DR+2*DST)){
    float a=0.f;
    for(int d=0;d<DIN;d++) a += xproj_w[t*DIN+d]*xv[d];
    dbc[t]=a;
  }
  __syncthreads();
  float a = dt_b[t];
  #pragma unroll
  for (int r=0;r<DR;r++) a += dt_w[t*DR+r]*dbc[r];
  delta[(size_t)blk*DIN+t] = softplusf_(a);
  if (t<DST){ Bc[(size_t)blk*DST+t]=dbc[DR+t]; Cc[(size_t)blk*DST+t]=dbc[DR+DST+t]; }
}

// ---------------------------------------------------------------------------
// Kernel 6: sequential selective scan. grid=B, block=256 (thread = channel d).
// ---------------------------------------------------------------------------
__global__ void __launch_bounds__(256) k_scan(const float* __restrict__ delta,
    const float* __restrict__ Bc, const float* __restrict__ Cc,
    const float* __restrict__ xc, const float* __restrict__ xz,
    const float* __restrict__ Alog, const float* __restrict__ Dp,
    float* __restrict__ ysb){
  int b = blockIdx.x; int d = threadIdx.x;
  float negA[DST], h[DST];
  #pragma unroll
  for(int n=0;n<DST;n++){ negA[n] = -__expf(Alog[d*DST+n]); h[n]=0.f; }
  float Dd = Dp[d];
  __shared__ float sB[DST], sC[DST];
  for (int l=0;l<LL;l++){
    size_t bl = (size_t)b*LL+l;
    if (d<DST) sB[d]=Bc[bl*DST+d];
    else if (d<2*DST) sC[d-DST]=Cc[bl*DST+(d-DST)];
    __syncthreads();
    float dl = delta[bl*DIN+d];
    float xv = xc[bl*DIN+d];
    float y = 0.f;
    #pragma unroll
    for (int n=0;n<DST;n++){
      float dA = __expf(dl*negA[n]);
      h[n] = dA*h[n] + dl*sB[n]*xv;
      y += h[n]*sC[n];
    }
    float z = xz[bl*(2*DIN) + DIN + d];
    ysb[bl*DIN+d] = (y + xv*Dd)*siluf_(z);
    __syncthreads();
  }
}

// ---------------------------------------------------------------------------
// Kernel 7: out-proj (256->128) + sigmoid gate with a_h/a_w + 2-layer MLP.
// Also writes sgate = sigmoid(a). grid=B*L, block=128.
// ---------------------------------------------------------------------------
__global__ void __launch_bounds__(128) k_out_gate_mlp(const float* __restrict__ ysb,
    const float* __restrict__ out_w, const float* __restrict__ abuf,
    const float* __restrict__ w1, const float* __restrict__ b1,
    const float* __restrict__ w2, const float* __restrict__ b2,
    float* __restrict__ sgate, float* __restrict__ outv){
  int blk=blockIdx.x; int b=blk/LL, l=blk%LL; int o=threadIdx.x;
  __shared__ float yv[DIN];
  __shared__ float vv[CC_];
  __shared__ float tv[CC_];
  yv[o] = ysb[(size_t)blk*DIN+o];
  yv[o+128] = ysb[(size_t)blk*DIN+o+128];
  __syncthreads();
  float m=0.f;
  for (int d=0;d<DIN;d++) m += out_w[o*DIN+d]*yv[d];
  size_t ai = ((size_t)b*CC_+o)*LL + l;
  float av = abuf[ai];
  sgate[ai] = sigmoidf_(av);
  vv[o] = sigmoidf_(m + av);
  __syncthreads();
  float a = b1[o];
  for (int c=0;c<CC_;c++) a += w1[o*CC_+c]*vv[c];
  tv[o] = geluf_(a);
  __syncthreads();
  float q = b2[o];
  for (int c=0;c<CC_;c++) q += w2[o*CC_+c]*tv[c];
  outv[ai] = q;
}

// ---------------------------------------------------------------------------
// Kernel 8: repack adjust_w (128x256 fp32) -> bf16 WMMA A-fragments.
// A-matrix 16x32 bf16 layout (ISA 7.12.2): lane<16 -> M=lane, K pairs in
// VGPR0..3 = K0..7, VGPR4..7 = K16..23; lane>=16 -> same M, K offset +8/+24.
// Fragment (mt,kt) stored contiguously: element index = lane*16 + j.
// ---------------------------------------------------------------------------
__global__ void __launch_bounds__(256) k_prepA(const float* __restrict__ adjust_w,
                                               bf16_t* __restrict__ Afrag){
  int frag = blockIdx.x; int mt=frag>>3, kt=frag&7;
  for (int e=threadIdx.x; e<512; e+=256){
    int l = e>>4, j = e&15;
    int hi = (l>=16)?1:0;
    int m  = mt*16 + (l&15);
    int pr = j>>1, bit=j&1;
    int base = (pr<4) ? (pr*2) : (16 + (pr-4)*2);
    int kl = base + bit + hi*8;
    int cc = kt*32 + kl;
    Afrag[(size_t)frag*512 + e] = (bf16_t)adjust_w[m*(2*CC_)+cc];
  }
}

// ---------------------------------------------------------------------------
// Kernel 9: fused final contraction + elementwise multiply with x.
//   out[b,o,h,w] = x[b,o,h,w] * ( A1@ (sh(:,h) .* Q(:,w)) + A2@ (sw(:,w) .* S(:,h)) )
// grid = B*H (one block per (b,h)); block = 256 = 8 waves; wave = one 16-row
// M-tile. 128x256x160 GEMM per block via v_wmma_f32_16x16x32_bf16.
// B-matrix 32x16 bf16 layout (wave32, ISA 7.12.4 mirrored): lane<16 holds
// K=0..7,16..23 (same order as A), lane>=16 holds K=8..15,24..31, N = lane&15.
// ---------------------------------------------------------------------------
__global__ void __launch_bounds__(256) k_final(const float* __restrict__ x,
    const bf16_t* __restrict__ Afrag,
    const float* __restrict__ sh, const float* __restrict__ Q,
    const float* __restrict__ sw, const float* __restrict__ S,
    float* __restrict__ out){
  int blk = blockIdx.x; int b = blk/HH, h = blk%HH;
  int tid = threadIdx.x, wv = tid>>5, lane = tid&31;
  __shared__ __attribute__((aligned(32))) bf16_t sfrag[8*512];
  __shared__ float s_sh[CC_];
  __shared__ float s_S[CC_];
  if (tid<CC_){
    s_sh[tid] = sh[((size_t)b*CC_+tid)*HH+h];
    s_S[tid]  = S [((size_t)b*CC_+tid)*HH+h];
  }
  __syncthreads();

  // A fragments for this wave's M-tile: resident in VGPRs for whole block.
  v16bf_t afr[8];
  const v16bf_t* Ap = (const v16bf_t*)Afrag;
  #pragma unroll
  for (int kt=0;kt<8;kt++) afr[kt] = Ap[((size_t)(wv*8+kt))*32 + lane];

  int n  = lane&15;
  int hi = lane>>4;

  for (int nt=0; nt<10; nt++){
    // Build B fragments (K=256 split into 8 steps of 32) for w-tile nt.
    for (int idx = tid; idx < 8*512; idx += 256){
      int kt = idx>>9; int e = idx&511;
      int l = e>>4, j = e&15;
      int nn = l&15;
      int w  = nt*16 + nn;
      int kl = j + ((j>=8)?8:0) + ((l>=16)?8:0);        // B layout: K = j<8 ? j+8hi : 16+8hi+(j-8)
      int cc = kt*32 + kl;
      float val;
      if (cc<CC_)      val = s_sh[cc]      * Q [((size_t)b*CC_+cc)*WW + w];
      else { int c2=cc-CC_; val = sw[((size_t)b*CC_+c2)*WW + w] * s_S[c2]; }
      sfrag[idx] = (bf16_t)val;
    }
    __syncthreads();

    v8f_t acc = {0.f,0.f,0.f,0.f,0.f,0.f,0.f,0.f};
    #pragma unroll
    for (int kt=0;kt<8;kt++){
      v16bf_t bfr = *(const v16bf_t*)(&sfrag[(kt<<9) + lane*16]);
      acc = __builtin_amdgcn_wmma_f32_16x16x32_bf16(
              /*neg_a=*/false, afr[kt],
              /*neg_b=*/false, bfr,
              /*c_mod=*/(short)0, acc,
              /*reuse_a=*/false, /*reuse_b=*/false);
    }

    // D layout: VGPR r, lanes 0-15 -> M=r, lanes 16-31 -> M=8+r; N=lane&15.
    int w = nt*16 + n;
    #pragma unroll
    for (int r=0;r<8;r++){
      int o = wv*16 + r + hi*8;
      size_t gi = (((size_t)b*CC_ + o)*HH + h)*WW + w;
      out[gi] = x[gi]*acc[r];
    }
    __syncthreads();
  }
}

// ---------------------------------------------------------------------------
extern "C" void kernel_launch(void* const* d_in, const int* in_sizes, int n_in,
                              void* d_out, int out_size, void* d_ws, size_t ws_size,
                              hipStream_t stream){
  (void)in_sizes; (void)n_in; (void)out_size; (void)ws_size;

  const float* x        = (const float*)d_in[0];
  const float* conv1_w  = (const float*)d_in[1];
  const float* conv1_b  = (const float*)d_in[2];
  const float* bn_g     = (const float*)d_in[3];
  const float* bn_b     = (const float*)d_in[4];
  const float* bn_m     = (const float*)d_in[5];
  const float* bn_v     = (const float*)d_in[6];
  const float* convh_w  = (const float*)d_in[7];
  const float* convh_b  = (const float*)d_in[8];
  const float* convw_w  = (const float*)d_in[9];
  const float* convw_b  = (const float*)d_in[10];
  const float* mlpx_w1  = (const float*)d_in[11];
  const float* mlpx_b1  = (const float*)d_in[12];
  const float* mlpx_w2  = (const float*)d_in[13];
  const float* mlpx_b2  = (const float*)d_in[14];
  const float* mlpy_w1  = (const float*)d_in[15];
  const float* mlpy_b1  = (const float*)d_in[16];
  const float* mlpy_w2  = (const float*)d_in[17];
  const float* mlpy_b2  = (const float*)d_in[18];
  const float* adjust_w = (const float*)d_in[19];
  const float* mx_in_w    = (const float*)d_in[20];
  const float* mx_conv_w  = (const float*)d_in[21];
  const float* mx_conv_b  = (const float*)d_in[22];
  const float* mx_xproj_w = (const float*)d_in[23];
  const float* mx_dt_w    = (const float*)d_in[24];
  const float* mx_dt_b    = (const float*)d_in[25];
  const float* mx_Alog    = (const float*)d_in[26];
  const float* mx_D       = (const float*)d_in[27];
  const float* mx_out_w   = (const float*)d_in[28];
  const float* my_in_w    = (const float*)d_in[29];
  const float* my_conv_w  = (const float*)d_in[30];
  const float* my_conv_b  = (const float*)d_in[31];
  const float* my_xproj_w = (const float*)d_in[32];
  const float* my_dt_w    = (const float*)d_in[33];
  const float* my_dt_b    = (const float*)d_in[34];
  const float* my_Alog    = (const float*)d_in[35];
  const float* my_D       = (const float*)d_in[36];
  const float* my_out_w   = (const float*)d_in[37];

  // ---- workspace carve-up (bytes) ----
  char* p = (char*)d_ws;
  bf16_t* Afrag = (bf16_t*)p; p += 64*512*sizeof(bf16_t);      // 64 KB
  const size_t NBC = (size_t)BB*CC_*HH;                        // 163840
  auto alloc = [&](size_t nfloats){ float* q=(float*)p; p += nfloats*sizeof(float); return q; };
  float* rowmean = alloc(NBC);
  float* colmean = alloc(NBC);
  float* a_h     = alloc(NBC);
  float* a_w     = alloc(NBC);
  float* shb     = alloc(NBC);      // sigmoid(a_h)
  float* swb     = alloc(NBC);      // sigmoid(a_w)
  float* Qb      = alloc(NBC);      // x_y after MLP
  float* Sb      = alloc(NBC);      // x_x after MLP
  float* xz      = alloc((size_t)BB*LL*2*DIN);
  float* xcb     = alloc((size_t)BB*LL*DIN);
  float* dlt     = alloc((size_t)BB*LL*DIN);
  float* Bcb     = alloc((size_t)BB*LL*DST);
  float* Ccb     = alloc((size_t)BB*LL*DST);
  float* ysb     = alloc((size_t)BB*LL*DIN);

  // Phase 0: pooled means (single full read of x).
  k_means<<<BB*CC_, 160, 0, stream>>>(x, rowmean, colmean);

  // Phase 1: tiny attention path (y, a_h, a_w) + WMMA A-operand repack.
  k_y_ahw<<<BB*(HH+WW), 128, 0, stream>>>(rowmean, colmean, conv1_w, conv1_b,
      bn_g, bn_b, bn_m, bn_v, convh_w, convh_b, convw_w, convw_b, a_h, a_w);
  k_prepA<<<64, 256, 0, stream>>>(adjust_w, Afrag);

  // Phase 2a: mamba along H (uses rowmean); result -> Sb, gates -> shb.
  k_inproj<<<BB*LL, 256, 0, stream>>>(rowmean, mx_in_w, xz);
  k_convsilu<<<BB*LL, 256, 0, stream>>>(xz, mx_conv_w, mx_conv_b, xcb);
  k_xproj_delta<<<BB*LL, 256, 0, stream>>>(xcb, mx_xproj_w, mx_dt_w, mx_dt_b, dlt, Bcb, Ccb);
  k_scan<<<BB, 256, 0, stream>>>(dlt, Bcb, Ccb, xcb, xz, mx_Alog, mx_D, ysb);
  k_out_gate_mlp<<<BB*LL, 128, 0, stream>>>(ysb, mx_out_w, a_h,
      mlpx_w1, mlpx_b1, mlpx_w2, mlpx_b2, shb, Sb);

  // Phase 2b: mamba along W (uses colmean); result -> Qb, gates -> swb.
  k_inproj<<<BB*LL, 256, 0, stream>>>(colmean, my_in_w, xz);
  k_convsilu<<<BB*LL, 256, 0, stream>>>(xz, my_conv_w, my_conv_b, xcb);
  k_xproj_delta<<<BB*LL, 256, 0, stream>>>(xcb, my_xproj_w, my_dt_w, my_dt_b, dlt, Bcb, Ccb);
  k_scan<<<BB, 256, 0, stream>>>(dlt, Bcb, Ccb, xcb, xz, my_Alog, my_D, ysb);
  k_out_gate_mlp<<<BB*LL, 128, 0, stream>>>(ysb, my_out_w, a_w,
      mlpy_w1, mlpy_b1, mlpy_w2, mlpy_b2, swb, Qb);

  // Phase 3: WMMA-fused final contraction + multiply by x (second x read + write).
  k_final<<<BB*HH, 256, 0, stream>>>(x, Afrag, shb, Qb, swb, Sb, (float*)d_out);
}
